// BaseNetwork_7825430413761
// MI455X (gfx1250) — compile-verified, matrix operand order checked by reading, not run on hardware
//
#include <hip/hip_runtime.h>
#include <hip/hip_bf16.h>

#define NG 2048
#define NODES 128
#define EPG 512            // edges per graph (128*4)
#define F_IN 64
#define DD 128
#define EPSV 1e-5f
#define SLOPE 0.01f
#define NEDGES (NG * EPG)

typedef __attribute__((ext_vector_type(16))) __bf16 v16bf;
typedef __attribute__((ext_vector_type(8)))  float  v8f;
typedef unsigned int u32x4 __attribute__((ext_vector_type(4)));
typedef int          i32x4 __attribute__((ext_vector_type(4)));
typedef int          i32x8 __attribute__((ext_vector_type(8)));

__device__ __forceinline__ v8f wmma_bf16(v16bf a, v16bf b, v8f c) {
  // (neg_a, A, neg_b, B, c_mod, C, reuse_a, reuse_b)
  return __builtin_amdgcn_wmma_f32_16x16x32_bf16(false, a, false, b, (short)0, c, false, false);
}

// A/B 16-bit fragment K mapping: lane half h=lane>>4; elem e<8 -> K=8h+e ; e>=8 -> K=16+8h+(e-8)

// value[lane&15][K] read row-contiguously from f32 LDS (A-matrix, or B from a transposed source)
__device__ __forceinline__ v16bf load_frag_rowK(const float* base, int ld, int rbase, int kbase, int lane) {
  const float* row = base + (rbase + (lane & 15)) * ld + kbase + ((lane >> 4) << 3);
  v16bf v;
#pragma unroll
  for (int i = 0; i < 8; ++i) {
    v[i]     = (__bf16)row[i];
    v[i + 8] = (__bf16)row[16 + i];
  }
  return v;
}

// value[K][lane&15] read column-strided from f32 LDS (B-matrix from row-major [K][N] source)
__device__ __forceinline__ v16bf load_frag_colK(const float* base, int ld, int kbase, int obase, int lane) {
  const float* p = base + (kbase + ((lane >> 4) << 3)) * ld + obase + (lane & 15);
  v16bf v;
#pragma unroll
  for (int i = 0; i < 8; ++i) {
    v[i]     = (__bf16)p[i * ld];
    v[i + 8] = (__bf16)p[(16 + i) * ld];
  }
  return v;
}

// C/D layout: elem r -> M = r + 8*(lane>>4), N = lane&15
__device__ __forceinline__ v8f load_cinit(const float* base, int ld, int mbase, int nbase, int lane) {
  const float* p = base + (mbase + ((lane >> 4) << 3)) * ld + nbase + (lane & 15);
  v8f c;
#pragma unroll
  for (int r = 0; r < 8; ++r) c[r] = p[r * ld];
  return c;
}

__device__ __forceinline__ void store_c(float* base, int ld, int mbase, int nbase, int lane, v8f c) {
  float* p = base + (mbase + ((lane >> 4) << 3)) * ld + nbase + (lane & 15);
#pragma unroll
  for (int r = 0; r < 8; ++r) p[r * ld] = c[r];
}

// pre-packed weight fragment: 16 contiguous bf16 per lane per (kt,nt) fragment
__device__ __forceinline__ v16bf load_wfrag(const __bf16* Wp, int kt, int nt, int lane) {
  return *(const v16bf*)(Wp + ((((kt << 3) + nt) * 32 + lane) << 4));
}

// ---------------- TDM: 2D tile global->LDS with optional LDS row padding ----------------
// tile_w dwords x tile_h rows, global row stride = stride_elems dwords.
// pad codes per D# group1: interval DWORDs = 2<<code ; amount DWORDs = code+1
__device__ __forceinline__ void tdm_load_2d(unsigned lds_byte_off, const void* gaddr,
                                            unsigned tile_w, unsigned tile_h, unsigned stride_elems,
                                            bool pad_en, unsigned pad_interval_code,
                                            unsigned pad_amount_code) {
  unsigned long long ga = (unsigned long long)gaddr;
  u32x4 g0;
  g0[0] = 1u;                                             // count=1, is_restore=0, gather off
  g0[1] = lds_byte_off;                                   // lds_addr
  g0[2] = (unsigned)(ga & 0xFFFFFFFFu);                   // global_addr[31:0]
  g0[3] = (unsigned)((ga >> 32) & 0x1FFFFFFu) | (2u << 30); // global_addr[56:32], type=2
  i32x8 g1;
  g1[0] = (int)((2u << 16) |                              // data_size = 4B
                (pad_en ? (1u << 20) : 0u) |
                (pad_interval_code << 22) | (pad_amount_code << 25));
  g1[1] = (int)((tile_w & 0xFFFFu) << 16);                // tensor_dim0[15:0] (abar addr = 0)
  g1[2] = (int)(((tile_w >> 16) & 0xFFFFu) | ((tile_h & 0xFFFFu) << 16)); // tensor_dim0 hi, tensor_dim1 lo
  g1[3] = (int)(((tile_h >> 16) & 0xFFFFu) | (tile_w << 16));             // tensor_dim1 hi, tile_dim0
  g1[4] = (int)(tile_h & 0xFFFFu);                        // tile_dim1 (tile_dim2 = 0)
  g1[5] = (int)stride_elems;                              // tensor_dim0_stride[31:0]
  g1[6] = 0;                                              // stride0 hi, stride1 lo
  g1[7] = 0;
  i32x4 z4 = {0, 0, 0, 0};
  i32x8 z8 = {0, 0, 0, 0, 0, 0, 0, 0};
  __builtin_amdgcn_tensor_load_to_lds(g0, g1, z4, z4, z8, 0);
}

// ---------------- weight pack kernel ----------------
// layout in ws (bf16): Wc0s[8192] Wc0n[8192] Wc1s[16384] Wc1n[16384] Wa[16384]
__global__ void pack_weights(const float* __restrict__ Wc0s, const float* __restrict__ Wc0n,
                             const float* __restrict__ Wc1s, const float* __restrict__ Wc1n,
                             const float* __restrict__ Wa, __bf16* __restrict__ out) {
  int tid = blockIdx.x * blockDim.x + threadIdx.x;
  const float* W;
  __bf16* dst;
  int idx;
  if (tid < 8192)        { W = Wc0s; dst = out;         idx = tid; }
  else if (tid < 16384)  { W = Wc0n; dst = out + 8192;  idx = tid - 8192; }
  else if (tid < 32768)  { W = Wc1s; dst = out + 16384; idx = tid - 16384; }
  else if (tid < 49152)  { W = Wc1n; dst = out + 32768; idx = tid - 32768; }
  else if (tid < 65536)  { W = Wa;   dst = out + 49152; idx = tid - 49152; }
  else return;
  int e    = idx & 15;
  int lane = (idx >> 4) & 31;
  int frag = idx >> 9;       // kt*8 + nt
  int nt = frag & 7, kt = frag >> 3;
  int half = lane >> 4;
  int kk  = e + 8 * half + ((e >= 8) ? 8 : 0);
  int row = kt * 32 + kk;
  int col = nt * 16 + (lane & 15);
  dst[idx] = (__bf16)W[row * DD + col];
}

// ---------------- fused conv GEMM: out = act(bn(A1@W1 + A2@W2 + bias)) * mw ----------------
template <int KTILES>
__device__ __forceinline__ void conv_gemm(const float* shA1, const float* shA2, float* shOut,
                                          const __bf16* W1p, const __bf16* W2p,
                                          const float* __restrict__ bias, const float* __restrict__ bg,
                                          const float* __restrict__ bb, const float* __restrict__ bm,
                                          const float* __restrict__ bv, const float* __restrict__ mono,
                                          int w, int lane) {
  int n0 = lane & 15, half = lane >> 4;
  v8f acc[8];
#pragma unroll
  for (int nt = 0; nt < 8; ++nt) {
    float bi = bias[nt * 16 + n0];
#pragma unroll
    for (int r = 0; r < 8; ++r) acc[nt][r] = bi;
  }
#pragma unroll
  for (int kt = 0; kt < KTILES; ++kt) {
    v16bf a1 = load_frag_rowK(shA1, DD, w * 16, kt * 32, lane);
    v16bf a2 = load_frag_rowK(shA2, DD, w * 16, kt * 32, lane);
#pragma unroll
    for (int nt = 0; nt < 8; ++nt) {
      acc[nt] = wmma_bf16(a1, load_wfrag(W1p, kt, nt, lane), acc[nt]);
      acc[nt] = wmma_bf16(a2, load_wfrag(W2p, kt, nt, lane), acc[nt]);
    }
  }
  float mw8[8];
#pragma unroll
  for (int r = 0; r < 8; ++r) mw8[r] = mono ? mono[w * 16 + half * 8 + r] : 1.0f;
#pragma unroll
  for (int nt = 0; nt < 8; ++nt) {
    int n = nt * 16 + n0;
    float sc = bg[n] * rsqrtf(bv[n] + EPSV);
    float sh = bb[n] - bm[n] * sc;
    float* p = shOut + (w * 16 + half * 8) * DD + n;
#pragma unroll
    for (int r = 0; r < 8; ++r) {
      float val = acc[nt][r] * sc + sh;
      val = val > 0.f ? val : SLOPE * val;
      p[r * DD] = val * mw8[r];
    }
  }
}

// LDS blob layout (single LDS object -> base offset 0)
#define OFF_H     0u
#define OFF_AGG   65536u
#define OFF_PAIR  131072u
#define OFF_DEG   147456u
#define OFF_SRC   147968u
#define OFF_DST   150016u
#define OFF_VEC   152064u
#define OFF_VEC2  152576u
#define SMEM_BYTES 153088u

// ---------------- one workgroup per graph: full fused pipeline ----------------
__global__ void __launch_bounds__(256)
gnn_graph_kernel(const float* __restrict__ x, const float* __restrict__ mono,
                 const __bf16* __restrict__ wpack,
                 const float* __restrict__ bc0, const float* __restrict__ g0,
                 const float* __restrict__ b0, const float* __restrict__ m0,
                 const float* __restrict__ v0,
                 const float* __restrict__ bc1, const float* __restrict__ g1,
                 const float* __restrict__ b1, const float* __restrict__ m1,
                 const float* __restrict__ v1,
                 const float* __restrict__ ba,
                 const float* __restrict__ Wr0, const float* __restrict__ br0,
                 const float* __restrict__ Wr1, const float* __restrict__ br1,
                 const float* __restrict__ Wout, const float* __restrict__ bout,
                 const int* __restrict__ edge_index, float* __restrict__ out) {
  __shared__ __align__(16) unsigned char smem[SMEM_BYTES];
  float* sh_h    = (float*)(smem + OFF_H);     // x -> conv0 -> conv1 -> newA/newB (in place)
  float* sh_agg  = (float*)(smem + OFF_AGG);   // agg0/agg1, then P = leaky(h@Wa+ba)
  float* sh_pair = (float*)(smem + OFF_PAIR);
  float* sh_deg  = (float*)(smem + OFF_DEG);
  int*   sh_src  = (int*)(smem + OFF_SRC);
  int*   sh_dst  = (int*)(smem + OFF_DST);
  float* sh_vec  = (float*)(smem + OFF_VEC);
  float* sh_vec2 = (float*)(smem + OFF_VEC2);

  const int g = blockIdx.x;
  const int tid = threadIdx.x;
  const int w = tid >> 5;         // wave id 0..7
  const int lane = tid & 31;

  const __bf16* W0s = wpack;
  const __bf16* W0n = wpack + 8192;
  const __bf16* W1s = wpack + 16384;
  const __bf16* W1n = wpack + 32768;
  const __bf16* Wap = wpack + 49152;

  const int* esrc = edge_index;
  const int* edst = edge_index + NEDGES;
  const float* monoG = mono + g * NODES;

  // -------- preamble --------
  // TDM: stage x[g] (128 rows x 64 f32) into sh_h with ld=128 via LDS padding:
  // after every 64 dwords (interval code 5) skip 64 dwords (amount code 63).
  if (w == 0) {
    tdm_load_2d(OFF_H, x + (size_t)g * NODES * F_IN,
                /*tile_w=*/F_IN, /*tile_h=*/NODES, /*stride=*/F_IN,
                /*pad_en=*/true, /*interval=*/5u, /*amount=*/63u);
    __builtin_amdgcn_s_wait_tensorcnt(0);
  }
  for (int i = tid; i < EPG; i += 256) {
    sh_src[i] = esrc[g * EPG + i] - g * NODES;
    sh_dst[i] = edst[g * EPG + i] - g * NODES;
  }
  for (int i = tid; i < NODES * DD; i += 256) sh_agg[i] = 0.f;
  if (tid < NODES) sh_deg[tid] = 0.f;
  __syncthreads();

  // -------- degree + agg0 (scatter-add over edges, F=64) --------
  for (int i = tid; i < EPG; i += 256) atomicAdd(&sh_deg[sh_dst[i]], 1.0f);
  for (int i = tid; i < EPG * F_IN; i += 256) {
    int e = i >> 6, f = i & 63;
    atomicAdd(&sh_agg[sh_dst[e] * DD + f], sh_h[sh_src[e] * DD + f]);
  }
  __syncthreads();
  for (int i = tid; i < NODES * F_IN; i += 256) {
    int n = i >> 6, f = i & 63;
    sh_agg[n * DD + f] /= fmaxf(sh_deg[n], 1.0f);
  }
  __syncthreads();

  // -------- conv0 + bn0 + leaky (in-place into sh_h, K=64) --------
  conv_gemm<2>(sh_h, sh_agg, sh_h, W0s, W0n, bc0, g0, b0, m0, v0, nullptr, w, lane);
  __syncthreads();

  // -------- agg1 (F=128) --------
  for (int i = tid; i < NODES * DD; i += 256) sh_agg[i] = 0.f;
  __syncthreads();
  for (int i = tid; i < EPG * DD; i += 256) {
    int e = i >> 7, f = i & 127;
    atomicAdd(&sh_agg[sh_dst[e] * DD + f], sh_h[sh_src[e] * DD + f]);
  }
  __syncthreads();
  for (int i = tid; i < NODES * DD; i += 256) {
    int n = i >> 7;
    sh_agg[i] /= fmaxf(sh_deg[n], 1.0f);
  }
  __syncthreads();

  // -------- conv1 + bn1 + leaky + monomer weight (in-place, K=128) --------
  conv_gemm<4>(sh_h, sh_agg, sh_h, W1s, W1n, bc1, g1, b1, m1, v1, monoG, w, lane);
  __syncthreads();

  // -------- P = leaky(h @ Wa + ba)  -> sh_agg  (pA rows 0..63, pB rows 64..127) --------
  {
    int n0 = lane & 15, half = lane >> 4;
    v8f acc[8];
#pragma unroll
    for (int nt = 0; nt < 8; ++nt) {
      float bi = ba[nt * 16 + n0];
#pragma unroll
      for (int r = 0; r < 8; ++r) acc[nt][r] = bi;
    }
#pragma unroll
    for (int kt = 0; kt < 4; ++kt) {
      v16bf a = load_frag_rowK(sh_h, DD, w * 16, kt * 32, lane);
#pragma unroll
      for (int nt = 0; nt < 8; ++nt)
        acc[nt] = wmma_bf16(a, load_wfrag(Wap, kt, nt, lane), acc[nt]);
    }
#pragma unroll
    for (int nt = 0; nt < 8; ++nt) {
      float* p = sh_agg + (w * 16 + half * 8) * DD + nt * 16 + n0;
#pragma unroll
      for (int r = 0; r < 8; ++r) {
        float val = acc[nt][r];
        p[r * DD] = val > 0.f ? val : SLOPE * val;
      }
    }
  }
  __syncthreads();

  // -------- pair = sigmoid(pA @ pB^T) : 64x64, 16 tiles over 8 waves --------
  {
#pragma unroll
    for (int i = 0; i < 2; ++i) {
      int t = w * 2 + i;
      int mt = t >> 2, nt = t & 3;
      v8f acc;
#pragma unroll
      for (int r = 0; r < 8; ++r) acc[r] = 0.f;
#pragma unroll
      for (int kt = 0; kt < 4; ++kt) {
        v16bf a = load_frag_rowK(sh_agg, DD, mt * 16, kt * 32, lane);           // pA[m][k]
        v16bf b = load_frag_rowK(sh_agg + 64 * DD, DD, nt * 16, kt * 32, lane); // pB[n][k] (B from transposed source)
        acc = wmma_bf16(a, b, acc);
      }
#pragma unroll
      for (int r = 0; r < 8; ++r) acc[r] = 1.0f / (1.0f + __expf(-acc[r]));
      store_c(sh_pair, 64, mt * 16, nt * 16, lane, acc);
    }
  }
  __syncthreads();

  // -------- newA = pair@pB + mA (waves 0..3) ; newB = pair^T@pA + mB (waves 4..7), in-place --------
  if (w < 4) {
    int mbase = w * 16;
    v8f acc[8];
#pragma unroll
    for (int nt = 0; nt < 8; ++nt) acc[nt] = load_cinit(sh_h, DD, mbase, nt * 16, lane);
#pragma unroll
    for (int kt = 0; kt < 2; ++kt) {
      v16bf a = load_frag_rowK(sh_pair, 64, mbase, kt * 32, lane);              // pair[m][k]
#pragma unroll
      for (int nt = 0; nt < 8; ++nt) {
        v16bf b = load_frag_colK(sh_agg + 64 * DD, DD, kt * 32, nt * 16, lane); // pB[k][n]
        acc[nt] = wmma_bf16(a, b, acc[nt]);
      }
    }
#pragma unroll
    for (int nt = 0; nt < 8; ++nt) store_c(sh_h, DD, mbase, nt * 16, lane, acc[nt]);
  } else {
    int m2 = (w - 4) * 16;
    v8f acc[8];
#pragma unroll
    for (int nt = 0; nt < 8; ++nt) acc[nt] = load_cinit(sh_h, DD, 64 + m2, nt * 16, lane);
#pragma unroll
    for (int kt = 0; kt < 2; ++kt) {
      v16bf a = load_frag_colK(sh_pair, 64, kt * 32, m2, lane);                 // pair^T[m][k] = pair[k][m]
#pragma unroll
      for (int nt = 0; nt < 8; ++nt) {
        v16bf b = load_frag_colK(sh_agg, DD, kt * 32, nt * 16, lane);           // pA[k][n]
        acc[nt] = wmma_bf16(a, b, acc[nt]);
      }
    }
#pragma unroll
    for (int nt = 0; nt < 8; ++nt) store_c(sh_h, DD, 64 + m2, nt * 16, lane, acc[nt]);
  }
  __syncthreads();

  // -------- mean pool over 128 nodes --------
  if (tid < DD) {
    float s = 0.f;
    for (int r = 0; r < NODES; ++r) s += sh_h[r * DD + tid];
    sh_vec[tid] = s * (1.0f / NODES);
  }
  __syncthreads();

  // -------- MLP head --------
  if (tid < DD) {
    float s = br0[tid];
    for (int k = 0; k < DD; ++k) s += sh_vec[k] * Wr0[k * DD + tid];
    sh_vec2[tid] = s > 0.f ? s : SLOPE * s;
  }
  __syncthreads();
  if (tid < DD) {
    float s = br1[tid];
    for (int k = 0; k < DD; ++k) s += sh_vec2[k] * Wr1[k * DD + tid];
    float r = s > 0.f ? s : SLOPE * s;
    sh_vec[tid] = r * Wout[tid];
  }
  __syncthreads();
  if (tid == 0) {
    float s = bout[0];
    for (int k = 0; k < DD; ++k) s += sh_vec[k];
    out[g] = s;
  }
}

extern "C" void kernel_launch(void* const* d_in, const int* in_sizes, int n_in,
                              void* d_out, int out_size, void* d_ws, size_t ws_size,
                              hipStream_t stream) {
  (void)in_sizes; (void)n_in; (void)out_size; (void)ws_size;
  const float* x    = (const float*)d_in[0];
  const float* mono = (const float*)d_in[1];
  const float* Wc0s = (const float*)d_in[2];
  const float* Wc0n = (const float*)d_in[3];
  const float* bc0  = (const float*)d_in[4];
  const float* g0   = (const float*)d_in[5];
  const float* b0   = (const float*)d_in[6];
  const float* m0   = (const float*)d_in[7];
  const float* v0   = (const float*)d_in[8];
  const float* Wc1s = (const float*)d_in[9];
  const float* Wc1n = (const float*)d_in[10];
  const float* bc1  = (const float*)d_in[11];
  const float* g1   = (const float*)d_in[12];
  const float* b1   = (const float*)d_in[13];
  const float* m1   = (const float*)d_in[14];
  const float* v1   = (const float*)d_in[15];
  const float* Wa   = (const float*)d_in[16];
  const float* ba   = (const float*)d_in[17];
  const float* Wr0  = (const float*)d_in[18];
  const float* br0  = (const float*)d_in[19];
  const float* Wr1  = (const float*)d_in[20];
  const float* br1  = (const float*)d_in[21];
  const float* Wout = (const float*)d_in[22];
  const float* bout = (const float*)d_in[23];
  const int* edge_index = (const int*)d_in[24];
  // d_in[25] = batch_index: unused (graphs are contiguous blocks of 128 nodes)

  __bf16* wpack = (__bf16*)d_ws;   // 65536 bf16 = 128KB
  pack_weights<<<256, 256, 0, stream>>>(Wc0s, Wc0n, Wc1s, Wc1n, Wa, wpack);
  gnn_graph_kernel<<<NG, 256, 0, stream>>>(
      x, mono, wpack,
      bc0, g0, b0, m0, v0,
      bc1, g1, b1, m1, v1,
      ba, Wr0, br0, Wr1, br1, Wout, bout,
      edge_index, (float*)d_out);
}
